// MLP2_47914655154752
// MI455X (gfx1250) — matrix operand order repaired and checked by reading, not compiled, not run on hardware
//
#include <hip/hip_runtime.h>
#include <hip/hip_bf16.h>

// ---------------------------------------------------------------------------
// MLP2 v4 (= v2 structure, reverted from the unroll-2 experiment that caused
// scratch spills + 6x code bloat): 17-layer MLP (ELU) fused with tensor-basis
// contraction. v_wmma_f32_16x16x32_f16 matmuls; 2 sample-tiles (32 samples)
// per wave so every B fragment loaded into VGPRs feeds 2 WMMAs (halves
// operand traffic, the binding constraint per roofline analysis).
// ---------------------------------------------------------------------------

typedef __attribute__((ext_vector_type(16))) _Float16 v16h;
typedef __attribute__((ext_vector_type(8)))  _Float16 v8h;
typedef __attribute__((ext_vector_type(8)))  float    v8f;

#define ACT_STRIDE 328                     // halves/row -> conflict-free b128 LDS
#define TILE_HALVES (16 * ACT_STRIDE)      // 5248 halves = 10496 B per tile
#define WAVES_PER_BLOCK 4
#define TILES_PER_WAVE 2
#define TILES_PER_BLOCK (WAVES_PER_BLOCK * TILES_PER_WAVE)

struct MlpParams {
    unsigned woff[17];   // half-element offsets into packed weights
    unsigned boff[17];   // float offsets into padded bias region
};

// ---------------------------------------------------------------------------
// Weight prepack: fp32 W[fanIn][fanOut] -> f16 B fragments, zero padded.
// Fragment f = nt*KT + kt covers K=[kt*32,+32) x N=[nt*16,+16).
// B layout (16-bit, 32x16): lanes 0-15 hold K=0..7,16..23 (N=lane&15), lanes 16-31
// hold K=8..15,24..31; half h within a lane is K = grp*8 + (h&7) + (h>>3)*16.
// Packed addr: frag*512 + lane*16 + h (each lane's 16 halves contiguous, 32B).
// ---------------------------------------------------------------------------
__global__ void pack_weights_kernel(const float* __restrict__ W,
                                    _Float16* __restrict__ dst,
                                    int fanIn, int fanOut, int KT, int NT)
{
    int t = blockIdx.x * blockDim.x + threadIdx.x;
    int total = KT * NT * 512;
    if (t >= total) return;
    int frag = t >> 9;
    int rem  = t & 511;
    int lane = rem >> 4;
    int h    = rem & 15;
    int nt = frag / KT;
    int kt = frag % KT;
    int n = nt * 16 + (lane & 15);
    int k = kt * 32 + (lane >> 4) * 8 + (h & 7) + (h >> 3) * 16;
    float v = 0.0f;
    if (k < fanIn && n < fanOut) v = W[(size_t)k * fanOut + n];
    dst[(size_t)frag * 512 + rem] = (_Float16)v;
}

// Bias prepack: fp32 -> fp32 zero-padded to NT*16 (unconditional loads later).
__global__ void pack_bias_kernel(const float* __restrict__ b,
                                 float* __restrict__ dst, int fanOut, int npad)
{
    for (int i = threadIdx.x; i < npad; i += blockDim.x)
        dst[i] = (i < fanOut) ? b[i] : 0.0f;
}

// ---------------------------------------------------------------------------
// One dense layer on this wave's TILES_PER_WAVE 16-sample tiles (LDS-resident
// activations, f16 [16][ACT_STRIDE] per tile). All A fragments are pulled into
// VGPRs first, so outputs overwrite the buffers in place; zero-padded weights
// neutralize stale padded columns. Each B fragment feeds TILES_PER_WAVE WMMAs.
// ---------------------------------------------------------------------------
template <int KIN, int NOUT, bool ELU>
__device__ __forceinline__ void mlp_layer(_Float16* __restrict__ actw,
                                          const _Float16* __restrict__ wpack,
                                          const float* __restrict__ bias,
                                          unsigned lane)
{
    constexpr int KT = (KIN + 31) / 32;
    constexpr int NT = (NOUT + 15) / 16;
    const unsigned m   = lane & 15u;
    const unsigned grp = lane >> 4;

    // A fragments: lane holds row m; halves 0-7 -> K = kt*32 + grp*8 + 0..7,
    // halves 8-15 -> K = kt*32 + 16 + grp*8 + 0..7 (two ds_load_b128 each).
    v16h a[TILES_PER_WAVE][KT];
#pragma unroll
    for (int t = 0; t < TILES_PER_WAVE; ++t) {
        const _Float16* base = actw + (size_t)t * TILE_HALVES
                             + (size_t)m * ACT_STRIDE + grp * 8;
#pragma unroll
        for (int kt = 0; kt < KT; ++kt) {
            v8h lo = *(const v8h*)(base + kt * 32);
            v8h hi = *(const v8h*)(base + kt * 32 + 16);
            a[t][kt] = __builtin_shufflevector(lo, hi, 0, 1, 2, 3, 4, 5, 6, 7,
                                               8, 9, 10, 11, 12, 13, 14, 15);
        }
    }

#pragma unroll 1
    for (int nt = 0; nt < NT; ++nt) {
        int n = nt * 16 + (int)m;
        float bv = bias[n];                       // padded: always in-bounds
        v8f c[TILES_PER_WAVE];
#pragma unroll
        for (int t = 0; t < TILES_PER_WAVE; ++t)
            c[t] = (v8f){bv, bv, bv, bv, bv, bv, bv, bv};
#pragma unroll
        for (int kt = 0; kt < KT; ++kt) {
            const _Float16* wp = wpack + ((size_t)(nt * KT + kt)) * 512 + lane * 16;
            v16h b = *(const v16h*)wp;            // 32B/lane, global_load_b128 x2
#pragma unroll
            for (int t = 0; t < TILES_PER_WAVE; ++t)
                c[t] = __builtin_amdgcn_wmma_f32_16x16x32_f16(
                           false, a[t][kt], false, b, (short)0, c[t], false, false);
        }
        // D layout: VGPR r holds (M = r + 8*grp, N = lane&15). ELU + f16 store.
#pragma unroll
        for (int t = 0; t < TILES_PER_WAVE; ++t) {
            _Float16* dst = actw + (size_t)t * TILE_HALVES;
#pragma unroll
            for (int r = 0; r < 8; ++r) {
                float v = c[t][r];
                if (ELU) v = (v > 0.0f) ? v : (__expf(v) - 1.0f);
                dst[(size_t)(r + 8 * grp) * ACT_STRIDE + n] = (_Float16)v;
            }
        }
    }
}

// Last layer (15 -> 10, no ELU): f32 result overlaid into the tile's own LDS
// region (A fragments are consumed into registers before the overwrite).
template <int KIN, int NOUT>
__device__ __forceinline__ void mlp_last(_Float16* __restrict__ act,
                                         float* __restrict__ cbuf,
                                         const _Float16* __restrict__ wpack,
                                         const float* __restrict__ bias,
                                         unsigned lane)
{
    constexpr int KT = (KIN + 31) / 32;
    const unsigned m   = lane & 15u;
    const unsigned grp = lane >> 4;

    v16h a[KT];
#pragma unroll
    for (int kt = 0; kt < KT; ++kt) {
        const _Float16* p = act + (size_t)m * ACT_STRIDE + kt * 32 + grp * 8;
        v8h lo = *(const v8h*)p;
        v8h hi = *(const v8h*)(p + 16);
        a[kt] = __builtin_shufflevector(lo, hi, 0, 1, 2, 3, 4, 5, 6, 7,
                                        8, 9, 10, 11, 12, 13, 14, 15);
    }
    int n = (int)m;
    float bv = bias[n];
    v8f c = {bv, bv, bv, bv, bv, bv, bv, bv};
#pragma unroll
    for (int kt = 0; kt < KT; ++kt) {
        const _Float16* wp = wpack + (size_t)kt * 512 + lane * 16;
        v16h b = *(const v16h*)wp;
        c = __builtin_amdgcn_wmma_f32_16x16x32_f16(
                false, a[kt], false, b, (short)0, c, false, false);
    }
#pragma unroll
    for (int r = 0; r < 8; ++r)
        cbuf[(size_t)(r + 8 * grp) * 16 + n] = c[r];
}

// ---------------------------------------------------------------------------
// Main kernel: 4 waves/block, 2 tiles (32 samples) per wave, barrier-free.
// ---------------------------------------------------------------------------
__global__ void __launch_bounds__(WAVES_PER_BLOCK * 32, 3)
mlp2_main_kernel(const float* __restrict__ x,
                 const float* __restrict__ y,
                 float* __restrict__ out,
                 const _Float16* __restrict__ wpack,
                 const float* __restrict__ bpack,
                 MlpParams P, int ntiles, int nsamples)
{
    __shared__ _Float16 smem[TILES_PER_BLOCK * TILE_HALVES];   // ~82 KB

    const unsigned tid  = threadIdx.x;
    const unsigned wave = tid >> 5;     // wave32
    const unsigned lane = tid & 31u;

    _Float16* actw = smem + (size_t)wave * TILES_PER_WAVE * TILE_HALVES;

    int tbase = blockIdx.x * TILES_PER_BLOCK + (int)wave * TILES_PER_WAVE;
    if (tbase >= ntiles) return;       // wave-uniform -> live waves keep EXEC all-1

    // Zero both activation tiles (finite padded columns guaranteed).
    {
        unsigned* a32 = (unsigned*)actw;
        for (int i = (int)lane; i < TILES_PER_WAVE * TILE_HALVES / 2; i += 32)
            a32[i] = 0u;
    }
    // Load x tiles: 16 samples x 5 invariants each, fp32 -> f16.
#pragma unroll
    for (int t = 0; t < TILES_PER_WAVE; ++t) {
        long base = (long)(tbase + t) * 16;
        for (int i = (int)lane; i < 16 * 5; i += 32) {
            int mm = i / 5, j = i % 5;
            long s = base + mm;
            float v = (s < nsamples) ? x[s * 5 + j] : 0.0f;
            actw[(size_t)t * TILE_HALVES + (size_t)mm * ACT_STRIDE + j] = (_Float16)v;
        }
    }

#define L(i, KIN, NOUT) \
    mlp_layer<KIN, NOUT, true>(actw, wpack + P.woff[i], bpack + P.boff[i], lane)
    L(0, 5, 15);    L(1, 15, 50);   L(2, 50, 50);   L(3, 50, 150);
    L(4, 150, 150); L(5, 150, 150); L(6, 150, 300); L(7, 300, 300);
    L(8, 300, 300); L(9, 300, 300); L(10, 300, 150); L(11, 150, 150);
    L(12, 150, 150); L(13, 150, 50); L(14, 50, 50);  L(15, 50, 15);
#undef L

    // Last layer + fused contraction per tile.
#pragma unroll
    for (int t = 0; t < TILES_PER_WAVE; ++t) {
        _Float16* act  = actw + (size_t)t * TILE_HALVES;
        float*    cbuf = (float*)act;            // overlay (A consumed to regs first)
        mlp_last<15, 10>(act, cbuf, wpack + P.woff[16], bpack + P.boff[16], lane);

        long base = (long)(tbase + t) * 16;
        for (int i = (int)lane; i < 16 * 9; i += 32) {
            int s = i / 9, cc = i % 9;
            long gs = base + s;
            if (gs >= nsamples) continue;
            const float* yp = y + gs * 90 + cc;
            float acc = 0.0f;
#pragma unroll
            for (int k = 0; k < 10; ++k) acc += cbuf[s * 16 + k] * yp[k * 9];
            out[gs * 9 + cc] = acc;
        }
    }
}

// ---------------------------------------------------------------------------
// Host launcher. d_in: x, y, then (W0,b0)..(W16,b16) fp32. d_out: [N,9] fp32.
// d_ws: packed f16 weights (~1.03 MB) + padded f32 biases (~10 KB).
// ---------------------------------------------------------------------------
extern "C" void kernel_launch(void* const* d_in, const int* in_sizes, int n_in,
                              void* d_out, int out_size, void* d_ws, size_t ws_size,
                              hipStream_t stream)
{
    static const int S[18] = {5, 15, 50, 50, 150, 150, 150, 300, 300, 300,
                              300, 150, 150, 150, 50, 50, 15, 10};

    const float* x = (const float*)d_in[0];
    const float* y = (const float*)d_in[1];

    // Workspace layout: [f16 weights | padded f32 biases]
    MlpParams P;
    unsigned woff = 0, bcnt = 0;
    for (int l = 0; l < 17; ++l) {
        int K = S[l], N = S[l + 1];
        int KT = (K + 31) / 32, NT = (N + 15) / 16;
        P.woff[l] = woff;  woff += (unsigned)(KT * NT * 512);
        P.boff[l] = bcnt;  bcnt += (unsigned)(NT * 16);
    }
    _Float16* wpack = (_Float16*)d_ws;
    float* bpack = (float*)(((uintptr_t)(wpack + woff) + 15) & ~(uintptr_t)15);

    for (int l = 0; l < 17; ++l) {
        int K = S[l], N = S[l + 1];
        int KT = (K + 31) / 32, NT = (N + 15) / 16;
        const float* W = (const float*)d_in[2 + 2 * l];
        const float* b = (const float*)d_in[3 + 2 * l];
        int total = KT * NT * 512;
        pack_weights_kernel<<<(total + 255) / 256, 256, 0, stream>>>(
            W, wpack + P.woff[l], K, N, KT, NT);
        pack_bias_kernel<<<1, 256, 0, stream>>>(b, bpack + P.boff[l], N, NT * 16);
    }

    int nsamples = in_sizes[0] / 5;
    int ntiles = (nsamples + 15) / 16;
    int blocks = (ntiles + TILES_PER_BLOCK - 1) / TILES_PER_BLOCK;
    mlp2_main_kernel<<<blocks, WAVES_PER_BLOCK * 32, 0, stream>>>(
        x, y, (float*)d_out, wpack, bpack, P, ntiles, nsamples);
}
